// SiReN_3401614098655
// MI455X (gfx1250) — matrix operand order repaired and checked by reading, not compiled, not run on hardware
//
#include <hip/hip_runtime.h>
#include <math.h>

#define NUM_U   100000
#define NUM_V   50000
#define N_NODES (NUM_U + NUM_V)
#define DIM     64
#define NTILES  (N_NODES / 16)   // 9375 exactly
#define REG_C   0.05f
#define GAMMA_C 1e-10f

typedef __attribute__((ext_vector_type(16))) _Float16 v16h;
typedef __attribute__((ext_vector_type(8)))  float    v8f;

// ---------------------------------------------------------------------------
// Small utility kernels
// ---------------------------------------------------------------------------
__global__ void zero_kernel(float* __restrict__ p, size_t n) {
  size_t t = (size_t)blockIdx.x * blockDim.x + threadIdx.x;
  if (t < n) p[t] = 0.f;
}

__global__ void copy2_kernel(float* __restrict__ a, float* __restrict__ b,
                             const float* __restrict__ s, size_t n) {
  size_t t = (size_t)blockIdx.x * blockDim.x + threadIdx.x;
  if (t < n) { float v = s[t]; a[t] = v; b[t] = v; }
}

__global__ void axpy_kernel(float* __restrict__ acc, const float* __restrict__ x, size_t n) {
  size_t t = (size_t)blockIdx.x * blockDim.x + threadIdx.x;
  if (t < n) acc[t] += x[t];
}

// ---------------------------------------------------------------------------
// Graph normalization: deg -> rsqrt -> per-edge weight
// ---------------------------------------------------------------------------
__global__ void deg_kernel(const long long* __restrict__ col, float* __restrict__ deg, long long E) {
  long long t = (long long)blockIdx.x * blockDim.x + threadIdx.x;
  if (t < E) atomicAdd(&deg[(int)col[t]], 1.0f);
}

__global__ void dinv_kernel(float* __restrict__ deg, int n) {
  int t = blockIdx.x * blockDim.x + threadIdx.x;
  if (t < n) { float d = deg[t]; deg[t] = (d > 0.f) ? rsqrtf(d) : 0.f; }
}

__global__ void ew_kernel(const long long* __restrict__ row, const long long* __restrict__ col,
                          const float* __restrict__ dinv, float* __restrict__ ew, long long E) {
  long long t = (long long)blockIdx.x * blockDim.x + threadIdx.x;
  if (t < E) ew[t] = dinv[(int)row[t]] * dinv[(int)col[t]];
}

// ---------------------------------------------------------------------------
// SPMM: out[row] += w[e] * x[col].  16 threads per edge, float4 per thread.
// ---------------------------------------------------------------------------
__global__ void spmm_kernel(const long long* __restrict__ row, const long long* __restrict__ col,
                            const float* __restrict__ ew,
                            const float* __restrict__ xin, float* __restrict__ xout,
                            long long E) {
  long long t = (long long)blockIdx.x * blockDim.x + threadIdx.x;
  if (t >= E * 16) return;
  long long e = t >> 4;
  int chunk = (int)(t & 15);
  int r = (int)row[e];
  int c = (int)col[e];
  float wgt = ew[e];
  const float4* src = (const float4*)(xin + (size_t)c * DIM);
  float4 v = src[chunk];
  float* dst = xout + (size_t)r * DIM + chunk * 4;
  atomicAdd(dst + 0, wgt * v.x);
  atomicAdd(dst + 1, wgt * v.y);
  atomicAdd(dst + 2, wgt * v.z);
  atomicAdd(dst + 3, wgt * v.w);
}

// ---------------------------------------------------------------------------
// Fused MLP + attention gates + softmax mix, all operands pre-swizzled into
// WMMA fragment layout in LDS (f16).  One 16-row tile per wave, 32 WMMAs/wave.
// Fragment layouts (ISA 7.12.2, wave32):
//   A 16x32 f16:  lanes 0-15: M=lane,   K = {0..7,16..23}
//                 lanes16-31: M=lane-16,K = {8..15,24..31}
//   B 32x16 f16:  lanes 0-15: N=lane,   K = {0..7,16..23} ; lanes16-31: N=lane-16, K = {8..15,24..31}
//   C/D f32 v8f:  elem r: (M = r + (lane<16?0:8), N = lane&15)
// ---------------------------------------------------------------------------
__global__ void __launch_bounds__(256) fused_attn_kernel(
    const float* __restrict__ emb_neg,
    const float* __restrict__ w0, const float* __restrict__ b0,
    const float* __restrict__ w1, const float* __restrict__ b1,
    const float* __restrict__ aw, const float* __restrict__ ab,
    const float* __restrict__ qw,
    float* __restrict__ acc) {
  __shared__ v16h wf0[8][32];            // [kt*4+nt][lane]  8KB
  __shared__ v16h wf1[8][32];            // 8KB
  __shared__ v16h wfa[8][32];            // 8KB
  __shared__ float b0s[DIM], b1s[DIM], abs_[DIM], qs[DIM];
  __shared__ v16h afrag[8][2][32];       // per wave: [kt][lane]  16KB
  __shared__ float gv[8][2][16];         // [wave][gp|gn][m]

  const int tid = threadIdx.x;

  // ---- swizzle weights into B-fragment layout (once per block) ----
  for (int i = tid; i < DIM * DIM; i += 256) {
    int k = i >> 6, n = i & 63;
    int kt = k >> 5, kk = k & 31;
    int nt = n >> 4;
    int fl = ((kk & 8) ? 16 : 0) + (n & 15);
    int e  = (kk & 7) + ((kk & 16) ? 8 : 0);
    int f  = kt * 4 + nt;
    ((_Float16*)&wf0[f][fl])[e] = (_Float16)w0[i];
    ((_Float16*)&wf1[f][fl])[e] = (_Float16)w1[i];
    ((_Float16*)&wfa[f][fl])[e] = (_Float16)aw[i];
  }
  if (tid < DIM) { b0s[tid] = b0[tid]; b1s[tid] = b1[tid]; abs_[tid] = ab[tid]; qs[tid] = qw[tid]; }
  __syncthreads();

  const int wv = tid >> 5;
  const int lane = tid & 31;
  const int tile = blockIdx.x * 8 + wv;
  const bool valid = (tile < NTILES);
  const int R = (valid ? tile : (NTILES - 1)) * 16;  // clamp stays inside same block
  const int mhi = (lane < 16) ? 0 : 8;
  const int ncol = lane & 15;
  v16h (*af)[32] = afrag[wv];

  // scatter one (m,n) f32 value into the f16 A-fragment buffer
  auto scatterA = [&](int idx, float v) {
    int m = idx >> 6, n = idx & 63;
    int kt = n >> 5, kk = n & 31;
    int half = (kk & 8) ? 16 : 0;
    int e = (kk & 7) + ((kk & 16) ? 8 : 0);
    ((_Float16*)&af[kt][half + m])[e] = (_Float16)v;
  };

  // D-position scatter constants (n = ncol + nt*16)
  const int halfD = (ncol & 8) ? 16 : 0;
  const int e0 = ncol & 7;

  v8f out[4];
  auto gemm = [&](const v16h (*wf)[32]) {
    v8f z = {};
#pragma unroll
    for (int nt = 0; nt < 4; ++nt) out[nt] = z;
#pragma unroll
    for (int kt = 0; kt < 2; ++kt) {
      v16h a = af[kt][lane];
#pragma unroll
      for (int nt = 0; nt < 4; ++nt)
        out[nt] = __builtin_amdgcn_wmma_f32_16x16x32_f16(
            false, a, false, wf[kt * 4 + nt][lane], (short)0, out[nt], false, false);
    }
  };

  // gate dot-product: g[m] = sum_n tanh(out + ab)[m][n] * q[n], shfl-reduced
  auto gate_dot = [&](int which) {
    float s[8];
#pragma unroll
    for (int r = 0; r < 8; ++r) s[r] = 0.f;
#pragma unroll
    for (int nt = 0; nt < 4; ++nt) {
      float q = qs[ncol + nt * 16];
      float bb = abs_[ncol + nt * 16];
#pragma unroll
      for (int r = 0; r < 8; ++r) s[r] += tanhf(out[nt][r] + bb) * q;
    }
#pragma unroll
    for (int off = 1; off < 16; off <<= 1)
#pragma unroll
      for (int r = 0; r < 8; ++r) s[r] += __shfl_xor(s[r], off, 16);
    if (ncol == 0) {
#pragma unroll
      for (int r = 0; r < 8; ++r) gv[wv][which][r + mhi] = s[r];
    }
  };

  // ---- layer 0: emb_neg tile -> A-frags ----
  for (int i = lane; i < 16 * DIM; i += 32) scatterA(i, emb_neg[(size_t)R * DIM + i]);
  __syncthreads();
  gemm(wf0);
  __syncthreads();
#pragma unroll
  for (int nt = 0; nt < 4; ++nt) {
    int kt = nt >> 1, e = e0 + ((nt & 1) ? 8 : 0);
    float bb = b0s[ncol + nt * 16];
#pragma unroll
    for (int r = 0; r < 8; ++r) {
      float v = out[nt][r] + bb;
      ((_Float16*)&af[kt][halfD + r + mhi])[e] = (_Float16)((v > 0.f) ? v : 0.f);
    }
  }
  __syncthreads();

  // ---- layer 1 -> z_n ----
  gemm(wf1);
  __syncthreads();
  v8f zn[4];
#pragma unroll
  for (int nt = 0; nt < 4; ++nt) {
    int kt = nt >> 1, e = e0 + ((nt & 1) ? 8 : 0);
    float bb = b1s[ncol + nt * 16];
#pragma unroll
    for (int r = 0; r < 8; ++r) {
      float v = out[nt][r] + bb;
      v = (v > 0.f) ? v : 0.f;
      zn[nt][r] = v;
      ((_Float16*)&af[kt][halfD + r + mhi])[e] = (_Float16)v;
    }
  }
  __syncthreads();

  // ---- gate(z_n) ----
  gemm(wfa);
  gate_dot(1);
  __syncthreads();

  // ---- z_p = acc/4 -> A-frags (+ keep f32 copy at D positions) ----
  for (int i = lane; i < 16 * DIM; i += 32) scatterA(i, 0.25f * acc[(size_t)R * DIM + i]);
  v8f zp[4];
#pragma unroll
  for (int nt = 0; nt < 4; ++nt)
#pragma unroll
    for (int r = 0; r < 8; ++r)
      zp[nt][r] = 0.25f * acc[(size_t)(R + r + mhi) * DIM + ncol + nt * 16];
  __syncthreads();

  // ---- gate(z_p) ----
  gemm(wfa);
  gate_dot(0);
  __syncthreads();

  // ---- softmax mix, store Z into acc (in place) ----
  if (valid) {
#pragma unroll
    for (int nt = 0; nt < 4; ++nt)
#pragma unroll
      for (int r = 0; r < 8; ++r) {
        int m = r + mhi;
        float gp = gv[wv][0][m], gn = gv[wv][1][m];
        float mx = fmaxf(gp, gn);
        float ep = expf(gp - mx), en = expf(gn - mx);
        float a0 = ep / (ep + en);
        acc[(size_t)(R + m) * DIM + ncol + nt * 16] =
            a0 * zp[nt][r] + (1.f - a0) * zn[nt][r];
      }
  }
}

// ---------------------------------------------------------------------------
// BPR loss: one thread per sample, block reduce, atomic into d_out
// ---------------------------------------------------------------------------
__global__ void __launch_bounds__(256) loss_kernel(
    const float* __restrict__ Z, const int* __restrict__ u,
    const int* __restrict__ iv, const int* __restrict__ jv,
    const float* __restrict__ sgn, float* __restrict__ out, int batch) {
  int tidx = blockIdx.x * 256 + threadIdx.x;
  float contrib = 0.f;
  if (tidx < batch) {
    const float4* up = (const float4*)(Z + (size_t)u[tidx] * DIM);
    const float4* ip = (const float4*)(Z + (size_t)(NUM_U + iv[tidx]) * DIM);
    const float4* jp = (const float4*)(Z + (size_t)(NUM_U + jv[tidx]) * DIM);
    float pos = 0.f, neg = 0.f, ru = 0.f, ri = 0.f, rj = 0.f;
#pragma unroll
    for (int k = 0; k < DIM / 4; ++k) {
      float4 a = up[k], b = ip[k], c = jp[k];
      pos += a.x * b.x + a.y * b.y + a.z * b.z + a.w * b.w;
      neg += a.x * c.x + a.y * c.y + a.z * c.z + a.w * c.w;
      ru  += a.x * a.x + a.y * a.y + a.z * a.z + a.w * a.w;
      ri  += b.x * b.x + b.y * b.y + b.z * b.z + b.w * b.w;
      rj  += c.x * c.x + c.y * c.y + c.z * c.z + c.w * c.w;
    }
    float x = sgn[tidx] * pos - neg;
    float s = 1.f / (1.f + expf(-x));
    float v = -logf(GAMMA_C + s);
    contrib = (v + REG_C * (ru + ri + rj)) / (float)batch;
  }
  __shared__ float red[256];
  red[threadIdx.x] = contrib;
  __syncthreads();
  for (int s2 = 128; s2 > 0; s2 >>= 1) {
    if (threadIdx.x < s2) red[threadIdx.x] += red[threadIdx.x + s2];
    __syncthreads();
  }
  if (threadIdx.x == 0) atomicAdd(out, red[0]);
}

// ---------------------------------------------------------------------------
// Host entry
// ---------------------------------------------------------------------------
extern "C" void kernel_launch(void* const* d_in, const int* in_sizes, int n_in,
                              void* d_out, int out_size, void* d_ws, size_t ws_size,
                              hipStream_t stream) {
  (void)n_in; (void)out_size; (void)ws_size;
  const float* emb_pos = (const float*)d_in[0];
  const float* emb_neg = (const float*)d_in[1];
  const float* w0 = (const float*)d_in[2];
  const float* b0 = (const float*)d_in[3];
  const float* w1 = (const float*)d_in[4];
  const float* b1 = (const float*)d_in[5];
  const float* aw = (const float*)d_in[6];
  const float* ab = (const float*)d_in[7];
  const float* qw = (const float*)d_in[8];
  const long long* eidx = (const long long*)d_in[9];
  const int* u  = (const int*)d_in[10];
  const int* iv = (const int*)d_in[11];
  const int* jv = (const int*)d_in[12];
  const float* sgn = (const float*)d_in[13];
  float* out = (float*)d_out;

  const long long E = (long long)in_sizes[9] / 2;
  const int batch = in_sizes[10];
  const size_t NODE_F = (size_t)N_NODES * DIM;

  char* ws = (char*)d_ws;
  auto alignup = [](size_t x) { return (x + 255) & ~(size_t)255; };
  size_t off = 0;
  float* deg = (float*)(ws + off); off = alignup(off + (size_t)N_NODES * 4);
  float* ew  = (float*)(ws + off); off = alignup(off + (size_t)E * 4);
  float* x0  = (float*)(ws + off); off = alignup(off + NODE_F * 4);
  float* x1  = (float*)(ws + off); off = alignup(off + NODE_F * 4);
  float* acc = (float*)(ws + off); off = alignup(off + NODE_F * 4);

  const long long* row = eidx;
  const long long* col = eidx + E;

  // degree -> dinv -> edge weights (reused for all 3 SPMM layers)
  zero_kernel<<<(N_NODES + 255) / 256, 256, 0, stream>>>(deg, (size_t)N_NODES);
  deg_kernel<<<(int)((E + 255) / 256), 256, 0, stream>>>(col, deg, E);
  dinv_kernel<<<(N_NODES + 255) / 256, 256, 0, stream>>>(deg, N_NODES);
  ew_kernel<<<(int)((E + 255) / 256), 256, 0, stream>>>(row, col, deg, ew, E);

  // acc = x0 = emb_pos
  copy2_kernel<<<(int)((NODE_F + 255) / 256), 256, 0, stream>>>(acc, x0, emb_pos, NODE_F);

  float* xin = x0; float* xout = x1;
  for (int l = 0; l < 3; ++l) {
    zero_kernel<<<(int)((NODE_F + 255) / 256), 256, 0, stream>>>(xout, NODE_F);
    long long tw = E * 16;
    spmm_kernel<<<(int)((tw + 255) / 256), 256, 0, stream>>>(row, col, ew, xin, xout, E);
    axpy_kernel<<<(int)((NODE_F + 255) / 256), 256, 0, stream>>>(acc, xout, NODE_F);
    float* tmp = xin; xin = xout; xout = tmp;
  }

  // MLP + attention gates + softmax mix (WMMA); Z overwrites acc in place
  fused_attn_kernel<<<(NTILES + 7) / 8, 256, 0, stream>>>(
      emb_neg, w0, b0, w1, b1, aw, ab, qw, acc);

  // loss
  zero_kernel<<<1, 256, 0, stream>>>(out, 1);
  loss_kernel<<<(batch + 255) / 256, 256, 0, stream>>>(acc, u, iv, jv, sgn, out, batch);
}
